// MORN_54709293416893
// MI455X (gfx1250) — matrix operand order repaired and checked by reading, not compiled, not run on hardware
//
#include <hip/hip_runtime.h>
#include <hip/hip_bf16.h>
#include <math.h>

typedef __attribute__((ext_vector_type(16))) __bf16 v16bf;
typedef __attribute__((ext_vector_type(8)))  float  v8f;

#define HID   128
#define NHEAD 4
#define DKH   32
#define NLAY  2
#define NPAT  2048
#define NWSI  2048
#define NGENE 100000
#define KPATCH 32
#define PDIM  1024
#define EBIG  500000
#define NOUTD 4

// ---------- helpers ----------

__device__ __forceinline__ unsigned f2ord(float f) {
  unsigned u = __float_as_uint(f);
  return (u & 0x80000000u) ? ~u : (u | 0x80000000u);
}
__device__ __forceinline__ float ord2f(unsigned u) {
  return __uint_as_float((u & 0x80000000u) ? (u & 0x7FFFFFFFu) : ~u);
}

// A fragment: 16x32 bf16 tile from f32 row-major A.
// Lane L (grp = L>>4, m = L&15) holds row m, K = kstep+grp*16 + {0..15}.
__device__ __forceinline__ v16bf load_a_frag(const float* __restrict__ A, int lda,
                                             int row0, int kstep, int lane) {
  const int m = lane & 15, grp = lane >> 4;
  const float* p = A + (size_t)(row0 + m) * lda + kstep + grp * 16;
  float4 c0 = *(const float4*)(p);
  float4 c1 = *(const float4*)(p + 4);
  float4 c2 = *(const float4*)(p + 8);
  float4 c3 = *(const float4*)(p + 12);
  v16bf a;
  a[0]  = (__bf16)c0.x; a[1]  = (__bf16)c0.y; a[2]  = (__bf16)c0.z; a[3]  = (__bf16)c0.w;
  a[4]  = (__bf16)c1.x; a[5]  = (__bf16)c1.y; a[6]  = (__bf16)c1.z; a[7]  = (__bf16)c1.w;
  a[8]  = (__bf16)c2.x; a[9]  = (__bf16)c2.y; a[10] = (__bf16)c2.z; a[11] = (__bf16)c2.w;
  a[12] = (__bf16)c3.x; a[13] = (__bf16)c3.y; a[14] = (__bf16)c3.z; a[15] = (__bf16)c3.w;
  return a;
}

// ---------- GEMM: C[M x 128] = act(oscale*(A[M x K] @ W[K x 128]) + bias) ----------
// Wt is pre-transposed bf16 (N-major: Wt[n*K + k]). 8 waves/block.
// 32 rows/block: each wave owns one 16-col tile and two 16-row tiles, so one
// shared B fragment feeds two WMMAs per K-step.
// B fragment: lane (col) holds contiguous K = kstep + grp*16 + 0..15.
__global__ __launch_bounds__(256)
void gemm_n128_wmma(const float* __restrict__ A, const __bf16* __restrict__ Wt,
                    const float* __restrict__ bias, float* __restrict__ C,
                    int M, int Kdim, int act, float oscale) {
  const int lane = threadIdx.x & 31;
  const int wid  = threadIdx.x >> 5;   // 0..7 -> N tile
  const int row0 = blockIdx.x * 32;
  const int col  = wid * 16 + (lane & 15);
  const int grp  = lane >> 4;
  v8f acc0 = {}, acc1 = {};
  for (int kstep = 0; kstep < Kdim; kstep += 32) {
    if (kstep + 32 < Kdim)
      __builtin_prefetch(A + (size_t)(row0 + (lane & 15)) * Kdim + kstep + 32 + grp * 8, 0, 1);
    v16bf a0 = load_a_frag(A, Kdim, row0,      kstep, lane);
    v16bf a1 = load_a_frag(A, Kdim, row0 + 16, kstep, lane);
    v16bf b  = *(const v16bf*)(Wt + (size_t)col * Kdim + kstep + grp * 16);
    acc0 = __builtin_amdgcn_wmma_f32_16x16x32_bf16(false, a0, false, b, (short)0, acc0,
                                                   false, false);
    acc1 = __builtin_amdgcn_wmma_f32_16x16x32_bf16(false, a1, false, b, (short)0, acc1,
                                                   false, false);
  }
  const float bv = bias ? bias[col] : 0.0f;
#pragma unroll
  for (int j = 0; j < 8; ++j) {            // C layout: VGPR j -> M = j + 8*grp
    int r = row0 + j + 8 * grp;
    float x0 = acc0[j] * oscale + bv;
    float x1 = acc1[j] * oscale + bv;
    if (act == 1) {
      x0 = 0.5f * x0 * (1.0f + erff(x0 * 0.70710678118f));
      x1 = 0.5f * x1 * (1.0f + erff(x1 * 0.70710678118f));
    }
    C[(size_t)r * HID + col]        = x0;
    C[(size_t)(r + 16) * HID + col] = x1;
  }
}

// ---------- per-head transform: Y[n,h,:] = X[n,h,:] @ R[h] (32x32 per head) ----------
// 8 waves/block: wave = head*2 + coltile; 16 rows/block; single K=32 WMMA per wave.
__global__ __launch_bounds__(256)
void headmm_wmma(const float* __restrict__ X, const float* __restrict__ R,
                 float* __restrict__ Y, int M) {
  const int lane = threadIdx.x & 31;
  const int wid  = threadIdx.x >> 5;
  const int hh   = wid >> 1;
  const int colh = (wid & 1) * 16 + (lane & 15);   // column within head: 0..31
  const int grp  = lane >> 4;
  const int row0 = blockIdx.x * 16;
  v16bf a = load_a_frag(X, HID, row0, hh * DKH, lane);
  const float* Rh = R + hh * DKH * DKH;
  v16bf b;
#pragma unroll
  for (int i = 0; i < 16; ++i)
    b[i] = (__bf16)Rh[(grp * 16 + i) * DKH + colh];
  v8f acc = {};
  acc = __builtin_amdgcn_wmma_f32_16x16x32_bf16(false, a, false, b, (short)0, acc,
                                                false, false);
#pragma unroll
  for (int j = 0; j < 8; ++j)
    Y[(size_t)(row0 + j + 8 * grp) * HID + hh * DKH + colh] = acc[j];
}

// ---------- weight convert+transpose: Wt[b][n][k] = bf16(W[b][k][n]) ----------
__global__ void convert_wt(const float* __restrict__ W, __bf16* __restrict__ Wt,
                           int K, int N, long total) {
  long idx = (long)blockIdx.x * blockDim.x + threadIdx.x;
  if (idx >= total) return;
  long kn = (long)K * N;
  long b = idx / kn, r = idx % kn;
  int k = (int)(r / N), n = (int)(r % N);
  Wt[b * kn + (long)n * K + k] = (__bf16)W[idx];
}

// ---------- masked mean pool over K patches ----------
__global__ __launch_bounds__(256)
void pool_kernel(const float* __restrict__ patches, const unsigned char* __restrict__ mask,
                 float* __restrict__ pooled) {
  int n = blockIdx.x, t = threadIdx.x;
  __shared__ float s_m[KPATCH];
  __shared__ float s_inv;
  if (t < KPATCH) s_m[t] = mask[n * KPATCH + t] ? 1.0f : 0.0f;
  __syncthreads();
  if (t == 0) {
    float d = 0.0f;
    for (int i = 0; i < KPATCH; ++i) d += s_m[i];
    s_inv = 1.0f / fmaxf(d, 1.0f);
  }
  __syncthreads();
  float inv = s_inv;
  for (int p = t; p < PDIM; p += 256) {
    float acc = 0.0f;
#pragma unroll
    for (int k = 0; k < KPATCH; ++k)
      acc += patches[((size_t)n * KPATCH + k) * PDIM + p] * s_m[k];
    pooled[(size_t)n * PDIM + p] = acc * inv;
  }
}

__global__ void gather_rows(const float* __restrict__ emb, const int* __restrict__ nid,
                            float* __restrict__ out, long n) {
  long idx = (long)blockIdx.x * blockDim.x + threadIdx.x;
  if (idx >= n) return;
  long r = idx >> 7; int c = (int)(idx & 127);
  out[idx] = emb[(size_t)nid[r] * HID + c];
}

__global__ void fill_f32(float* p, float v, long n) {
  long i = (long)blockIdx.x * blockDim.x + threadIdx.x;
  if (i < n) p[i] = v;
}
__global__ void fill_u32(unsigned* p, unsigned v, long n) {
  long i = (long)blockIdx.x * blockDim.x + threadIdx.x;
  if (i < n) p[i] = v;
}

// ---------- edge kernels ----------
__global__ void edge_logit_max(const float* __restrict__ ke, const float* __restrict__ q,
                               const int* __restrict__ src, const int* __restrict__ dst,
                               const float* __restrict__ pri, float* __restrict__ logit,
                               unsigned* __restrict__ segmax, int E, float scale) {
  int idx = blockIdx.x * blockDim.x + threadIdx.x;
  if (idx >= E * NHEAD) return;
  int e = idx >> 2, h = idx & 3;
  int s = src[e], d = dst[e];
  const float4* kp = (const float4*)(ke + (size_t)s * HID + h * DKH);
  const float4* qp = (const float4*)(q  + (size_t)d * HID + h * DKH);
  float acc = 0.0f;
#pragma unroll
  for (int i = 0; i < 8; ++i) {
    float4 a = kp[i], b = qp[i];
    acc += a.x * b.x + a.y * b.y + a.z * b.z + a.w * b.w;
  }
  float l = acc * pri[h] * scale;
  logit[idx] = l;
  atomicMax(&segmax[(size_t)d * NHEAD + h], f2ord(l));
}

__global__ void edge_exp_sum(float* __restrict__ logit, const int* __restrict__ dst,
                             const unsigned* __restrict__ segmax, float* __restrict__ segsum,
                             int E) {
  int idx = blockIdx.x * blockDim.x + threadIdx.x;
  if (idx >= E * NHEAD) return;
  int e = idx >> 2, h = idx & 3;
  int d = dst[e];
  float m = ord2f(segmax[(size_t)d * NHEAD + h]);
  float ex = __expf(logit[idx] - m);
  logit[idx] = ex;
  atomicAdd(&segsum[(size_t)d * NHEAD + h], ex);
}

__global__ void edge_scatter(const float* __restrict__ logit, const float* __restrict__ ve,
                             const int* __restrict__ src, const int* __restrict__ dst,
                             const float* __restrict__ w, const float* __restrict__ segsum,
                             float* __restrict__ agg, int E) {
  long idx = (long)blockIdx.x * blockDim.x + threadIdx.x;
  if (idx >= (long)E * HID) return;
  int c = (int)(idx & (HID - 1));
  long e = idx >> 7;
  int h = c >> 5;
  int d = dst[e], s = src[e];
  float attn = logit[e * NHEAD + h] / segsum[(size_t)d * NHEAD + h] * w[e];
  atomicAdd(&agg[(size_t)d * HID + c], ve[(size_t)s * HID + c] * attn);
}

// ---------- skip-blend + layernorm (one 128-thread block per row) ----------
__global__ __launch_bounds__(128)
void blend_ln(const float* __restrict__ trans, const float* __restrict__ hin,
              const float* __restrict__ skipv, const float* __restrict__ g,
              const float* __restrict__ bta, float* __restrict__ hout) {
  int row = blockIdx.x, t = threadIdx.x;
  __shared__ float red[HID];
  float a = 1.0f / (1.0f + __expf(-skipv[0]));
  float x = trans[(size_t)row * HID + t] * a + hin[(size_t)row * HID + t] * (1.0f - a);
  red[t] = x; __syncthreads();
  for (int s = 64; s > 0; s >>= 1) { if (t < s) red[t] += red[t + s]; __syncthreads(); }
  float mu = red[0] * (1.0f / HID);
  __syncthreads();
  float d = x - mu;
  red[t] = d * d; __syncthreads();
  for (int s = 64; s > 0; s >>= 1) { if (t < s) red[t] += red[t + s]; __syncthreads(); }
  float var = red[0] * (1.0f / HID);
  hout[(size_t)row * HID + t] = d * rsqrtf(var + 1e-5f) * g[t] + bta[t];
}

// ---------- final projection (128 -> 4) ----------
__global__ void out_proj(const float* __restrict__ h0, const float* __restrict__ W,
                         const float* __restrict__ b, float* __restrict__ out) {
  int idx = blockIdx.x * blockDim.x + threadIdx.x;
  if (idx >= NPAT * NOUTD) return;
  int r = idx >> 2, o = idx & 3;
  float acc = b[o];
  const float* hp = h0 + (size_t)r * HID;
#pragma unroll 8
  for (int k = 0; k < HID; ++k) acc += hp[k] * W[k * NOUTD + o];
  out[idx] = acc;
}

// ---------- host orchestration ----------

static inline int gdiv(long n, int b) { return (int)((n + b - 1) / b); }

extern "C" void kernel_launch(void* const* d_in, const int* in_sizes, int n_in,
                              void* d_out, int out_size, void* d_ws, size_t ws_size,
                              hipStream_t stream) {
  (void)in_sizes; (void)n_in; (void)out_size; (void)ws_size;
  const float* wsi_patches = (const float*)d_in[0];
  const float* emb_patient = (const float*)d_in[1];
  const float* emb_gene    = (const float*)d_in[2];
  const float* wsi_proj_w  = (const float*)d_in[3];
  const float* wsi_proj_b  = (const float*)d_in[4];
  const float* kW = (const float*)d_in[5];
  const float* qW = (const float*)d_in[6];
  const float* vW = (const float*)d_in[7];
  const float* aW = (const float*)d_in[8];
  const float* kb = (const float*)d_in[9];
  const float* qb = (const float*)d_in[10];
  const float* vb = (const float*)d_in[11];
  const float* ab = (const float*)d_in[12];
  const float* rel_att = (const float*)d_in[13];
  const float* rel_msg = (const float*)d_in[14];
  const float* rel_pri = (const float*)d_in[15];
  const float* skip    = (const float*)d_in[16];
  const float* ln_scale = (const float*)d_in[17];
  const float* ln_bias  = (const float*)d_in[18];
  const float* out_w = (const float*)d_in[19];
  const float* out_b = (const float*)d_in[20];
  const float* wE[4] = { (const float*)d_in[21], (const float*)d_in[22],
                         (const float*)d_in[23], (const float*)d_in[24] };
  const int* nid_patient = (const int*)d_in[25];
  const int* nid_gene    = (const int*)d_in[26];
  const int* srcE[4] = { (const int*)d_in[27], (const int*)d_in[29],
                         (const int*)d_in[31], (const int*)d_in[33] };
  const int* dstE[4] = { (const int*)d_in[28], (const int*)d_in[30],
                         (const int*)d_in[32], (const int*)d_in[34] };
  const unsigned char* wsi_patch_mask = (const unsigned char*)d_in[35];

  const int Ns[3] = { NPAT, NWSI, NGENE };
  const int sT[4] = { 1, 2, 0, 0 };
  const int dT[4] = { 0, 0, 2, 1 };
  const int nE[4] = { NWSI, EBIG, EBIG, NPAT };
  const float cnt[3] = { 2.0f, 1.0f, 1.0f };

  // ---- workspace carve (256B aligned) ----
  char* ws = (char*)d_ws;
  size_t off = 0;
  auto alloc = [&](size_t bytes) -> void* {
    void* p = ws + off;
    off = (off + bytes + 255) & ~(size_t)255;
    return p;
  };
  float* pooled = (float*)alloc((size_t)NWSI * PDIM * 4);
  float *hA[3], *hB[3], *kbuf[3], *qbuf[3], *vbuf[3], *aggb[3];
  for (int t = 0; t < 3; ++t) hA[t]   = (float*)alloc((size_t)Ns[t] * HID * 4);
  for (int t = 0; t < 3; ++t) hB[t]   = (float*)alloc((size_t)Ns[t] * HID * 4);
  for (int t = 0; t < 3; ++t) kbuf[t] = (float*)alloc((size_t)Ns[t] * HID * 4);
  for (int t = 0; t < 3; ++t) qbuf[t] = (float*)alloc((size_t)Ns[t] * HID * 4);
  for (int t = 0; t < 3; ++t) vbuf[t] = (float*)alloc((size_t)Ns[t] * HID * 4);
  for (int t = 0; t < 3; ++t) aggb[t] = (float*)alloc((size_t)Ns[t] * HID * 4);
  float* keb = (float*)alloc((size_t)NGENE * HID * 4);
  float* veb = (float*)alloc((size_t)NGENE * HID * 4);
  float* trans = keb;  // alias: ke scratch is dead by the time trans is needed
  float*    logit  = (float*)   alloc((size_t)EBIG * NHEAD * 4);
  unsigned* segmax = (unsigned*)alloc((size_t)NGENE * NHEAD * 4);
  float*    segsum = (float*)   alloc((size_t)NGENE * NHEAD * 4);
  __bf16* projWt = (__bf16*)alloc((size_t)HID * PDIM * 2);
  __bf16* kWt = (__bf16*)alloc((size_t)NLAY * 3 * HID * HID * 2);
  __bf16* qWt = (__bf16*)alloc((size_t)NLAY * 3 * HID * HID * 2);
  __bf16* vWt = (__bf16*)alloc((size_t)NLAY * 3 * HID * HID * 2);
  __bf16* aWt = (__bf16*)alloc((size_t)NLAY * 3 * HID * HID * 2);

  // ---- weights -> bf16 transposed ----
  { long tot = (long)PDIM * HID;
    convert_wt<<<gdiv(tot,256),256,0,stream>>>(wsi_proj_w, projWt, PDIM, HID, tot); }
  { long tot = (long)NLAY * 3 * HID * HID;
    convert_wt<<<gdiv(tot,256),256,0,stream>>>(kW, kWt, HID, HID, tot);
    convert_wt<<<gdiv(tot,256),256,0,stream>>>(qW, qWt, HID, HID, tot);
    convert_wt<<<gdiv(tot,256),256,0,stream>>>(vW, vWt, HID, HID, tot);
    convert_wt<<<gdiv(tot,256),256,0,stream>>>(aW, aWt, HID, HID, tot); }

  // ---- init node features ----
  gather_rows<<<gdiv((long)NPAT * HID,256),256,0,stream>>>(emb_patient, nid_patient, hA[0], (long)NPAT * HID);
  gather_rows<<<gdiv((long)NGENE * HID,256),256,0,stream>>>(emb_gene, nid_gene, hA[2], (long)NGENE * HID);
  pool_kernel<<<NWSI,256,0,stream>>>(wsi_patches, wsi_patch_mask, pooled);
  gemm_n128_wmma<<<NWSI/32,256,0,stream>>>(pooled, projWt, wsi_proj_b, hA[1],
                                           NWSI, PDIM, /*gelu=*/1, 1.0f);

  float* hcur[3] = { hA[0], hA[1], hA[2] };
  float* hnxt[3] = { hB[0], hB[1], hB[2] };
  const float scale = 0.17677669529f;  // 1/sqrt(32)

  for (int L = 0; L < NLAY; ++L) {
    for (int t = 0; t < 3; ++t) {
      size_t wo = (size_t)(L * 3 + t) * HID * HID;
      size_t bo = (size_t)(L * 3 + t) * HID;
      gemm_n128_wmma<<<Ns[t]/32,256,0,stream>>>(hcur[t], kWt + wo, kb + bo, kbuf[t], Ns[t], HID, 0, 1.0f);
      gemm_n128_wmma<<<Ns[t]/32,256,0,stream>>>(hcur[t], qWt + wo, qb + bo, qbuf[t], Ns[t], HID, 0, 1.0f);
      gemm_n128_wmma<<<Ns[t]/32,256,0,stream>>>(hcur[t], vWt + wo, vb + bo, vbuf[t], Ns[t], HID, 0, 1.0f);
      fill_f32<<<gdiv((long)Ns[t] * HID,256),256,0,stream>>>(aggb[t], 0.0f, (long)Ns[t] * HID);
    }
    for (int e = 0; e < 4; ++e) {
      int Nsrc = Ns[sT[e]], Ndst = Ns[dT[e]], E = nE[e];
      const float* Ratt = rel_att + (size_t)(L * 4 + e) * NHEAD * DKH * DKH;
      const float* Rmsg = rel_msg + (size_t)(L * 4 + e) * NHEAD * DKH * DKH;
      const float* pri  = rel_pri + (size_t)(L * 4 + e) * NHEAD;
      headmm_wmma<<<Nsrc/16,256,0,stream>>>(kbuf[sT[e]], Ratt, keb, Nsrc);
      headmm_wmma<<<Nsrc/16,256,0,stream>>>(vbuf[sT[e]], Rmsg, veb, Nsrc);
      fill_u32<<<gdiv((long)Ndst * NHEAD,256),256,0,stream>>>(segmax, 0x007FFFFFu, (long)Ndst * NHEAD);
      fill_f32<<<gdiv((long)Ndst * NHEAD,256),256,0,stream>>>(segsum, 0.0f, (long)Ndst * NHEAD);
      edge_logit_max<<<gdiv((long)E * NHEAD,256),256,0,stream>>>(
          keb, qbuf[dT[e]], srcE[e], dstE[e], pri, logit, segmax, E, scale);
      edge_exp_sum<<<gdiv((long)E * NHEAD,256),256,0,stream>>>(logit, dstE[e], segmax, segsum, E);
      edge_scatter<<<gdiv((long)E * HID,256),256,0,stream>>>(
          logit, veb, srcE[e], dstE[e], wE[e], segsum, aggb[dT[e]], E);
    }
    for (int t = 0; t < 3; ++t) {
      size_t wo = (size_t)(L * 3 + t) * HID * HID;
      size_t bo = (size_t)(L * 3 + t) * HID;
      gemm_n128_wmma<<<Ns[t]/32,256,0,stream>>>(aggb[t], aWt + wo, ab + bo, trans,
                                                Ns[t], HID, 0, 1.0f / cnt[t]);
      blend_ln<<<Ns[t],128,0,stream>>>(trans, hcur[t], skip + (size_t)L * 3 + t,
                                       ln_scale + bo, ln_bias + bo, hnxt[t]);
    }
    for (int t = 0; t < 3; ++t) { float* tmp = hcur[t]; hcur[t] = hnxt[t]; hnxt[t] = tmp; }
  }

  out_proj<<<gdiv((long)NPAT * NOUTD,256),256,0,stream>>>(hcur[0], out_w, out_b, (float*)d_out);
}
